// PropModule_72619307041247
// MI455X (gfx1250) — compile-verified, matrix operand order checked by reading, not run on hardware
//
#include <hip/hip_runtime.h>
#include <hip/hip_bf16.h>
#include <stdint.h>

// ---------- types ----------
typedef __bf16 bf16_t;
typedef __attribute__((ext_vector_type(16))) __bf16        v16bf;
typedef __attribute__((ext_vector_type(8)))  float         v8f;
typedef __attribute__((ext_vector_type(4)))  unsigned int  u32x4;
typedef __attribute__((ext_vector_type(4)))  int           i32x4;
typedef __attribute__((ext_vector_type(8)))  int           i32x8;

union Frag32B { v16bf v; u32x4 u[2]; };

// problem constants
static constexpr int BB = 2;
static constexpr int NN = 1024;
static constexpr int RR = 16384;

// ---------- TDM availability ----------
#if __has_builtin(__builtin_amdgcn_tensor_load_to_lds) && __has_builtin(__builtin_amdgcn_s_wait_tensorcnt)
#define HAVE_TDM 1
#else
#define HAVE_TDM 0
#endif

#if HAVE_TDM
// Issue one 2D TDM tile load: rows of `tile_d1` x `tile_d0` bf16 elements,
// row stride `stride` (elements). tensor_d0/d1 give the remaining in-bounds
// extent so the TDM zero-fills out-of-range reads (handles ragged K/N tiles).
__device__ __forceinline__ void tdm_load_2d(unsigned ldsaddr, const bf16_t* g,
                                            unsigned tile_d0, unsigned tile_d1,
                                            unsigned long long stride,
                                            unsigned tensor_d0, unsigned tensor_d1)
{
    unsigned long long ga = (unsigned long long)(size_t)g;
    u32x4 g0;
    g0.x = 1u;                                   // count=1 (valid descriptor)
    g0.y = ldsaddr;                              // LDS byte address
    g0.z = (unsigned)ga;                         // global_addr[31:0]
    g0.w = (unsigned)(ga >> 32) | (2u << 30);    // global_addr[56:32] | type=2
    i32x8 g1;
    g1[0] = (int)(1u << 16);                                   // wg_mask=0, data_size=1 (2B)
    g1[1] = (int)((tensor_d0 & 0xFFFFu) << 16);                // abar=0 | tensor_dim0 lo
    g1[2] = (int)((tensor_d0 >> 16) | ((tensor_d1 & 0xFFFFu) << 16));
    g1[3] = (int)((tensor_d1 >> 16) | (tile_d0 << 16));        // tensor_dim1 hi | tile_dim0
    g1[4] = (int)(tile_d1 & 0xFFFFu);                          // tile_dim1 | tile_dim2=0
    g1[5] = (int)(unsigned)(stride & 0xFFFFFFFFull);           // tensor_dim0_stride lo
    g1[6] = (int)(unsigned)((stride >> 32) & 0xFFFFull);       // stride hi | dim1_stride lo=0
    g1[7] = 0;
    i32x4 z4 = {0, 0, 0, 0};
#if __clang_major__ >= 23
    i32x8 z8 = {0, 0, 0, 0, 0, 0, 0, 0};
    __builtin_amdgcn_tensor_load_to_lds(g0, g1, z4, z4, z8, 0);
#else
    __builtin_amdgcn_tensor_load_to_lds(g0, g1, z4, z4, 0);
#endif
}
#endif

// ---------- elementwise helper kernels ----------
__global__ __launch_bounds__(256) void k_f32_to_bf16(const float* __restrict__ x,
                                                     bf16_t* __restrict__ y, long n) {
    long i = (long)blockIdx.x * blockDim.x + threadIdx.x;
    if (i < n) y[i] = (bf16_t)x[i];
}

__global__ __launch_bounds__(256) void k_fill0_bf16(bf16_t* __restrict__ y, long n) {
    long i = (long)blockIdx.x * blockDim.x + threadIdx.x;
    if (i < n) y[i] = (bf16_t)0.f;
}

// tiled transpose+convert: in f32 [Z][H][W] -> out bf16 [Z][W][H]
__global__ __launch_bounds__(256) void k_transpose_f32_bf16(
    const float* __restrict__ in, bf16_t* __restrict__ out, int H, int W)
{
    __shared__ float tile[32][33];
    long base = (long)blockIdx.z * H * W;
    int h0 = blockIdx.y * 32, w0 = blockIdx.x * 32;
    int tx = threadIdx.x & 31, ty = threadIdx.x >> 5;     // 32x8
#pragma unroll
    for (int j = 0; j < 32; j += 8) {
        int h = h0 + ty + j, w = w0 + tx;
        tile[ty + j][tx] = (h < H && w < W) ? in[base + (long)h * W + w] : 0.f;
    }
    __syncthreads();
#pragma unroll
    for (int j = 0; j < 32; j += 8) {
        int w = w0 + ty + j, h = h0 + tx;
        if (w < W && h < H) out[base + (long)w * H + h] = (bf16_t)tile[tx][ty + j];
    }
}

// Ra [B,R,8] (f32) -> relin[b,r,64..71] (bf16), relin row stride 72
__global__ __launch_bounds__(256) void k_scatter_ra(const float* __restrict__ ra,
                                                    bf16_t* __restrict__ relin) {
    long total = (long)BB * RR * 8;
    long i = (long)blockIdx.x * blockDim.x + threadIdx.x;
    if (i >= total) return;
    int  c  = (int)(i & 7);
    long br = i >> 3;
    relin[br * 72 + 64 + c] = (bf16_t)ra[i];
}

// ---------- WMMA GEMM (all operands pre-laid-out; no in-loop transpose) ----------
// C[M,N] = A[M,K] * Bt[N,K]^T (+bias) (+relu); optional transposed copy Ct[N,M].
// block = 256 thr = 8 waves (4x2); block tile 128(M) x 64(N); wave tile 32x32;
// K step 64; LDS staged by TDM (double buffered) or vector loads (fallback).
template <bool RELU, bool OUTBF16>
__global__ __launch_bounds__(256) void gemm_bt_wmma(
    const bf16_t* __restrict__ A, int lda, long sA,
    const bf16_t* __restrict__ Bt, int ldbt, long sBt,
    const float* __restrict__ bias,
    void* __restrict__ Cp, int ldc, long sC,
    bf16_t* __restrict__ Ct, int ldct, long sCt,
    int M, int N, int K)
{
    __shared__ __attribute__((aligned(16))) bf16_t lA[2][128 * 64];  // [m][k]
    __shared__ __attribute__((aligned(16))) bf16_t lB[2][64 * 64];   // [n][k]

    const int b = blockIdx.z;
    const bf16_t* Ab = A  + (long)b * sA;
    const bf16_t* Bb = Bt + (long)b * sBt;
    const int t = threadIdx.x, wave = t >> 5, lane = t & 31;
    const int wm = wave >> 1, wn = wave & 1;
    const int m0 = blockIdx.y * 128;
    const int n0 = blockIdx.x * 64;

    v8f accs[2][2];
#pragma unroll
    for (int i = 0; i < 2; ++i)
#pragma unroll
        for (int j = 0; j < 2; ++j) accs[i][j] = (v8f){0.f,0.f,0.f,0.f,0.f,0.f,0.f,0.f};

    auto compute = [&](const bf16_t* __restrict__ pA, const bf16_t* __restrict__ pB) {
#pragma unroll
        for (int kh = 0; kh < 2; ++kh) {
            Frag32B a[2], bb[2];
            const int off0 = kh * 64 + (lane >> 4) * 16;   // byte offset in 128B row
#pragma unroll
            for (int i = 0; i < 2; ++i) {
                const char* pa = (const char*)(pA + (wm * 32 + i * 16 + (lane & 15)) * 64);
                a[i].u[0] = *(const u32x4*)(pa + off0);
                a[i].u[1] = *(const u32x4*)(pa + off0 + 32);
                const char* pb = (const char*)(pB + (wn * 32 + i * 16 + (lane & 15)) * 64);
                bb[i].u[0] = *(const u32x4*)(pb + off0);
                bb[i].u[1] = *(const u32x4*)(pb + off0 + 32);
            }
#pragma unroll
            for (int tm = 0; tm < 2; ++tm)
#pragma unroll
                for (int tn = 0; tn < 2; ++tn)
                    accs[tm][tn] = __builtin_amdgcn_wmma_f32_16x16x32_bf16(
                        false, a[tm].v, false, bb[tn].v, (short)0, accs[tm][tn], false, false);
        }
    };

#if HAVE_TDM
    const unsigned ldsA0 = (unsigned)(size_t)&lA[0][0];
    const unsigned ldsA1 = (unsigned)(size_t)&lA[1][0];
    const unsigned ldsB0 = (unsigned)(size_t)&lB[0][0];
    const unsigned ldsB1 = (unsigned)(size_t)&lB[1][0];
    auto issue = [&](int bufi, int k0) {
        tdm_load_2d(bufi ? ldsA1 : ldsA0, Ab + (long)m0 * lda + k0, 64u, 128u,
                    (unsigned long long)lda, (unsigned)(K - k0), (unsigned)(M - m0));
        tdm_load_2d(bufi ? ldsB1 : ldsB0, Bb + (long)n0 * ldbt + k0, 64u, 64u,
                    (unsigned long long)ldbt, (unsigned)(K - k0), (unsigned)(N - n0));
    };
    int buf = 0;
    if (wave == 0) issue(0, 0);
    for (int k0 = 0; k0 < K; k0 += 64) {
        const bool nxt = (k0 + 64) < K;
        if (wave == 0) {
            if (nxt) {
                issue(buf ^ 1, k0 + 64);
                __builtin_amdgcn_s_wait_tensorcnt(2);   // oldest pair (current buf) done
            } else {
                __builtin_amdgcn_s_wait_tensorcnt(0);
            }
        }
        __syncthreads();                 // publish TDM-filled LDS to all waves
        compute(&lA[buf][0], &lB[buf][0]);
        __syncthreads();                 // buf may be overwritten next iteration
        buf ^= 1;
    }
#else
    for (int k0 = 0; k0 < K; k0 += 64) {
        // A tile: 128x64 elems as 4096 dword chunks
        for (int c = t; c < 4096; c += 256) {
            int row = c >> 5, k2 = (c & 31) * 2, gk = k0 + k2;
            const bf16_t* src = Ab + (long)(m0 + row) * lda + gk;
            unsigned v = 0;
            if (gk + 1 < K) v = *(const unsigned*)src;
            else if (gk < K) { unsigned short h; __builtin_memcpy(&h, src, 2); v = h; }
            *(unsigned*)&lA[0][row * 64 + k2] = v;
        }
        // B tile: 64x64 elems as 2048 dword chunks
        for (int c = t; c < 2048; c += 256) {
            int row = c >> 5, k2 = (c & 31) * 2, gk = k0 + k2;
            unsigned v = 0;
            if (n0 + row < N) {
                const bf16_t* src = Bb + (long)(n0 + row) * ldbt + gk;
                if (gk + 1 < K) v = *(const unsigned*)src;
                else if (gk < K) { unsigned short h; __builtin_memcpy(&h, src, 2); v = h; }
            }
            *(unsigned*)&lB[0][row * 64 + k2] = v;
        }
        __syncthreads();
        compute(&lA[0][0], &lB[0][0]);
        __syncthreads();
    }
#endif

    // ---- epilogue: lane L, vgpr i -> (m = i + 8*(L>=16), n = L&15) ----
    const int nn = lane & 15, mb = (lane >> 4) * 8;
#pragma unroll
    for (int tm = 0; tm < 2; ++tm)
#pragma unroll
        for (int tn = 0; tn < 2; ++tn) {
            int gn = n0 + wn * 32 + tn * 16 + nn;
            if (gn >= N) continue;
            float bv = bias ? bias[gn] : 0.f;
#pragma unroll
            for (int i = 0; i < 8; ++i) {
                long gm = (long)m0 + wm * 32 + tm * 16 + mb + i;
                float v = accs[tm][tn][i] + bv;
                if (RELU) v = v > 0.f ? v : 0.f;
                if (OUTBF16) ((bf16_t*)Cp)[(long)b * sC + gm * ldc + gn] = (bf16_t)v;
                else         ((float*)Cp)[(long)b * sC + gm * ldc + gn] = v;
                if (Ct) Ct[(long)b * sCt + (long)gn * ldct + gm] = (bf16_t)v;
            }
        }
}

// ---------- host-side helpers ----------
static void run_gemm(hipStream_t s, bool relu, bool obf,
                     const bf16_t* A, int lda, long sA,
                     const bf16_t* Bt, int ldbt, long sBt,
                     const float* bias, void* C, int ldc, long sC,
                     bf16_t* Ct, int ldct, long sCt,
                     int M, int Nn, int K)
{
    dim3 g((Nn + 63) / 64, (M + 127) / 128, BB), blk(256);
    if (relu)
        gemm_bt_wmma<true, true><<<g, blk, 0, s>>>(A, lda, sA, Bt, ldbt, sBt, bias, C, ldc, sC, Ct, ldct, sCt, M, Nn, K);
    else if (obf)
        gemm_bt_wmma<false, true><<<g, blk, 0, s>>>(A, lda, sA, Bt, ldbt, sBt, bias, C, ldc, sC, Ct, ldct, sCt, M, Nn, K);
    else
        gemm_bt_wmma<false, false><<<g, blk, 0, s>>>(A, lda, sA, Bt, ldbt, sBt, bias, C, ldc, sC, Ct, ldct, sCt, M, Nn, K);
}

static void conv(hipStream_t s, const float* x, bf16_t* y, long n) {
    k_f32_to_bf16<<<dim3((unsigned)((n + 255) / 256)), 256, 0, s>>>(x, y, n);
}
static void trconv(hipStream_t s, const float* in, bf16_t* out, int H, int W, int Z) {
    dim3 g((W + 31) / 32, (H + 31) / 32, Z);
    k_transpose_f32_bf16<<<g, 256, 0, s>>>(in, out, H, W);
}

extern "C" void kernel_launch(void* const* d_in, const int* in_sizes, int n_in,
                              void* d_out, int out_size, void* d_ws, size_t ws_size,
                              hipStream_t stream)
{
    const float* state = (const float*)d_in[0];
    const float* Rr    = (const float*)d_in[1];
    const float* Rs    = (const float*)d_in[2];
    const float* Ra    = (const float*)d_in[3];
    // d_in[4] = pstep (device scalar) -> fixed 3 for graph determinism
    const float* We1 = (const float*)d_in[5];  const float* be1 = (const float*)d_in[6];
    const float* We2 = (const float*)d_in[7];  const float* be2 = (const float*)d_in[8];
    const float* Wr1 = (const float*)d_in[9];  const float* br1 = (const float*)d_in[10];
    const float* Wr2 = (const float*)d_in[11]; const float* br2 = (const float*)d_in[12];
    const float* Wr3 = (const float*)d_in[13]; const float* br3 = (const float*)d_in[14];
    const float* Wrp = (const float*)d_in[15]; const float* brp = (const float*)d_in[16];
    const float* Wpp = (const float*)d_in[17]; const float* bpp = (const float*)d_in[18];
    const float* Wq1 = (const float*)d_in[19]; const float* bq1 = (const float*)d_in[20];
    const float* Wq2 = (const float*)d_in[21]; const float* bq2 = (const float*)d_in[22];

    size_t off = 0;
    auto carve = [&](long elems) -> bf16_t* {
        bf16_t* p = (bf16_t*)((char*)d_ws + off);
        off += ((size_t)elems * sizeof(bf16_t) + 255) & ~(size_t)255;
        return p;
    };
    const long nR = (long)BB * NN * RR;
    bf16_t* Rrb    = carve(nR);                  // [B][N][R]  (A of agg)
    bf16_t* Rrt    = carve(nR);                  // [B][R][N]  (A of state_r / Er)
    bf16_t* Rst    = carve(nR);                  // [B][R][N]  (A of state_s / Es)
    bf16_t* stateb = carve((long)BB * NN * 32);  // [B][N][32]
    bf16_t* statet = carve((long)BB * 32 * NN);  // [B][32][N] (Bt of state_r/s)
    bf16_t* We1t = carve(128 * 32);   bf16_t* We2t = carve(128 * 128);
    bf16_t* Wr1t = carve(128 * 72);   bf16_t* Wr2t = carve(128 * 128);
    bf16_t* Wr3t = carve(128 * 128);  bf16_t* Wrpt = carve(128 * 384);
    bf16_t* Wppt = carve(128 * 256);  bf16_t* Wq1t = carve(128 * 128);
    bf16_t* Wq2t = carve(3 * 128);
    bf16_t* hp1     = carve((long)BB * NN * 128);
    bf16_t* pcat    = carve((long)BB * NN * 256);  // [particle_encode | agg]
    bf16_t* relin   = carve((long)BB * RR * 72);   // [state_r | state_s | Ra]
    bf16_t* hr1     = carve((long)BB * RR * 128);
    bf16_t* hr2     = carve((long)BB * RR * 128);
    bf16_t* relcat  = carve((long)BB * RR * 384);  // [relation_encode | Er | Es]
    bf16_t* releff  = carve((long)BB * RR * 128);
    bf16_t* relefft = carve((long)BB * 128 * RR);  // Bt of agg
    bf16_t* E       = carve((long)BB * NN * 128);
    bf16_t* Et      = carve((long)BB * 128 * NN);  // Bt of Er/Es
    bf16_t* hq      = carve((long)BB * NN * 128);

    // ---- conversions (one pass; bf16 Rr-family stays L2-resident after) ----
    conv(stream, Rr, Rrb, nR);
    trconv(stream, Rr, Rrt, NN, RR, BB);
    trconv(stream, Rs, Rst, NN, RR, BB);
    conv(stream, state, stateb, (long)BB * NN * 32);
    trconv(stream, state, statet, NN, 32, BB);
    trconv(stream, We1, We1t, 32, 128, 1);   trconv(stream, We2, We2t, 128, 128, 1);
    trconv(stream, Wr1, Wr1t, 72, 128, 1);   trconv(stream, Wr2, Wr2t, 128, 128, 1);
    trconv(stream, Wr3, Wr3t, 128, 128, 1);  trconv(stream, Wrp, Wrpt, 384, 128, 1);
    trconv(stream, Wpp, Wppt, 256, 128, 1);  trconv(stream, Wq1, Wq1t, 128, 128, 1);
    trconv(stream, Wq2, Wq2t, 128, 3, 1);
    {
        long n = (long)BB * RR * 8;
        k_scatter_ra<<<dim3((unsigned)((n + 255) / 256)), 256, 0, stream>>>(Ra, relin);
    }
    k_fill0_bf16<<<dim3((unsigned)(((long)BB * NN * 128 + 255) / 256)), 256, 0, stream>>>(E,  (long)BB * NN * 128);
    k_fill0_bf16<<<dim3((unsigned)(((long)BB * NN * 128 + 255) / 256)), 256, 0, stream>>>(Et, (long)BB * NN * 128);

    const long sR    = (long)NN * RR;
    const long sSt   = (long)NN * 32;
    const long sRel  = (long)RR * 128;
    const long sRin  = (long)RR * 72;
    const long sRcat = (long)RR * 384;
    const long sP    = (long)NN * 128;
    const long sPcat = (long)NN * 256;

    // ---- particle encoder -> pcat[:,0:128] ----
    run_gemm(stream, true, true, stateb, 32, sSt, We1t, 32, 0, be1,
             hp1, 128, sP, nullptr, 0, 0, NN, 128, 32);
    run_gemm(stream, true, true, hp1, 128, sP, We2t, 128, 0, be2,
             pcat, 256, sPcat, nullptr, 0, 0, NN, 128, 128);

    // ---- state_r / state_s -> relin[:,0:32] / [:,32:64] ----
    run_gemm(stream, false, true, Rrt, NN, sR, statet, NN, sSt, nullptr,
             relin, 72, sRin, nullptr, 0, 0, RR, 32, NN);
    run_gemm(stream, false, true, Rst, NN, sR, statet, NN, sSt, nullptr,
             relin + 32, 72, sRin, nullptr, 0, 0, RR, 32, NN);

    // ---- relation encoder -> relcat[:,0:128] ----
    run_gemm(stream, true, true, relin, 72, sRin, Wr1t, 72, 0, br1,
             hr1, 128, sRel, nullptr, 0, 0, RR, 128, 72);
    run_gemm(stream, true, true, hr1, 128, sRel, Wr2t, 128, 0, br2,
             hr2, 128, sRel, nullptr, 0, 0, RR, 128, 128);
    run_gemm(stream, true, true, hr2, 128, sRel, Wr3t, 128, 0, br3,
             relcat, 384, sRcat, nullptr, 0, 0, RR, 128, 128);

    // ---- propagation loop (pstep = 3) ----
    for (int step = 0; step < 3; ++step) {
        run_gemm(stream, false, true, Rrt, NN, sR, Et, NN, sP, nullptr,
                 relcat + 128, 384, sRcat, nullptr, 0, 0, RR, 128, NN);
        run_gemm(stream, false, true, Rst, NN, sR, Et, NN, sP, nullptr,
                 relcat + 256, 384, sRcat, nullptr, 0, 0, RR, 128, NN);
        // rel_eff (also emit transposed copy for the scatter GEMM)
        run_gemm(stream, true, true, relcat, 384, sRcat, Wrpt, 384, 0, brp,
                 releff, 128, sRel, relefft, RR, sRel, RR, 128, 384);
        // agg = Rr @ rel_eff -> pcat[:,128:256]
        run_gemm(stream, false, true, Rrb, RR, sR, relefft, RR, sRel, nullptr,
                 pcat + 128, 256, sPcat, nullptr, 0, 0, NN, 128, RR);
        // E = relu(pcat @ Wpp + bpp)  (also emit E^T for next step's Er/Es)
        run_gemm(stream, true, true, pcat, 256, sPcat, Wppt, 256, 0, bpp,
                 E, 128, sP, Et, NN, sP, NN, 128, 256);
    }

    // ---- predictor -> d_out [B,N,3] f32 ----
    run_gemm(stream, true, true, E, 128, sP, Wq1t, 128, 0, bq1,
             hq, 128, sP, nullptr, 0, 0, NN, 128, 128);
    run_gemm(stream, false, false, hq, 128, sP, Wq2t, 128, 0, bq2,
             d_out, 3, (long)NN * 3, nullptr, 0, 0, NN, 3, 128);
}
